// Disentangler_83133386981634
// MI455X (gfx1250) — matrix operand order repaired and checked by reading, not compiled, not run on hardware
//
#include <hip/hip_runtime.h>

// ---------------- problem constants ----------------
#define T_    16
#define TOK_  4096
#define D_    2048
#define C_    1024
#define H_    2048
#define NMLP  32          // 16 node + 16 edge MLPs
#define LSAMP 1000
#define LPAD  1024
// ---------------- tiling ----------------
#define LB    128         // rows (L) per workgroup tile
#define NB    128         // cols (N) per workgroup tile
#define KB    32          // K step (matches WMMA bf16 K=32)
#define BST   40          // LDS row stride (bf16 elems): 80B, 16B-aligned
#define CST   (NB + 8)    // LDS row stride for epilogue transpose buffer

typedef __attribute__((ext_vector_type(16))) __bf16 v16bf;
typedef __attribute__((ext_vector_type(2)))  __bf16 v2bf;
typedef __attribute__((ext_vector_type(8)))  float  v8f;

union Frag16 { v16bf v; __bf16 e[16]; unsigned u[8]; uint4 q[2]; };
union F8     { float4 q[2]; float f[8]; };

__device__ __forceinline__ float gelu_exact(float v) {
    return 0.5f * v * (1.0f + erff(v * 0.70710678118654752f));
}

// Packed f32x2 -> bf16x2 conversion: single v_cvt_pk_bf16_f32 when available.
__device__ __forceinline__ unsigned pack_bf16(float lo, float hi) {
#if __has_builtin(__builtin_amdgcn_cvt_pk_bf16_f32)
    union { v2bf v; unsigned u; } p;
    p.v = __builtin_amdgcn_cvt_pk_bf16_f32(lo, hi);
    return p.u;
#else
    union { __bf16 h[2]; unsigned u; } p;
    p.h[0] = (__bf16)lo;
    p.h[1] = (__bf16)hi;
    return p.u;
#endif
}

// ---- B tile: register stage (k-pair x 8 h) -------------------------------
struct BStage { F8 r0, r1; };

__device__ __forceinline__ void load_btile(const float* __restrict__ Wk, int ldw,
                                           int ncol0, int kt, int kp, int hb,
                                           BStage& s) {
    const float* p0 = Wk + (size_t)(kt * KB + kp) * ldw + ncol0 + hb;
    const float* p1 = p0 + ldw;
    s.r0.q[0] = ((const float4*)p0)[0];
    s.r0.q[1] = ((const float4*)p0)[1];
    s.r1.q[0] = ((const float4*)p1)[0];
    s.r1.q[1] = ((const float4*)p1)[1];
}

// XOR-swizzled transposed store: 8 packed b32 stores, ~4-way max conflict.
__device__ __forceinline__ void store_btile(__bf16* Bt, int kp, int hb,
                                            const BStage& s) {
#pragma unroll
    for (int i = 0; i < 8; ++i) {
        int h   = hb + i;
        int blk = (kp >> 3) ^ ((h >> 3) & 3);          // swizzle 8-elem blocks
        int off = h * BST + blk * 8 + (kp & 7);        // even -> 4B aligned
        *(unsigned*)&Bt[off] = pack_bf16(s.r0.f[i], s.r1.f[i]);
    }
}

// B fragment: two aligned b128 LDS loads with matching de-swizzle.
__device__ __forceinline__ void load_bfrag(const __bf16* Bt, int h, int kg,
                                           Frag16& f) {
    int sw = (h >> 3) & 3;
    f.q[0] = *(const uint4*)&Bt[h * BST + ((kg + 0) ^ sw) * 8];
    f.q[1] = *(const uint4*)&Bt[h * BST + ((kg + 2) ^ sw) * 8];
}

// A fragment, f32 source with convert (GEMM1 gather path)
__device__ __forceinline__ void load_afrag_f32(const float* __restrict__ row,
                                               bool valid, int kt, int ca,
                                               Frag16& f) {
    if (valid) {
        const float* p = row + kt * KB;
        F8 u0, u1;
        u0.q[0] = *(const float4*)(p + ca);
        u0.q[1] = *(const float4*)(p + ca + 4);
        u1.q[0] = *(const float4*)(p + 16 + ca);
        u1.q[1] = *(const float4*)(p + 16 + ca + 4);
#pragma unroll
        for (int i = 0; i < 4; ++i) {
            f.u[i]     = pack_bf16(u0.f[2 * i], u0.f[2 * i + 1]);
            f.u[4 + i] = pack_bf16(u1.f[2 * i], u1.f[2 * i + 1]);
        }
    } else {
        f.q[0] = make_uint4(0, 0, 0, 0);
        f.q[1] = make_uint4(0, 0, 0, 0);
    }
}

// A fragment, bf16 source (GEMM2: h buffer is already fragment-friendly)
__device__ __forceinline__ void load_afrag_bf16(const __bf16* __restrict__ row,
                                                int kt, int ca, Frag16& f) {
    const __bf16* p = row + kt * KB;
    f.q[0] = *(const uint4*)(p + ca);
    f.q[1] = *(const uint4*)(p + 16 + ca);
}

__device__ __forceinline__ void do_wmma(const Frag16 a[4], const __bf16* Bt,
                                        int wn, int ncol, int kg, v8f acc[4][2]) {
    Frag16 b0, b1;
    load_bfrag(Bt, wn * 32 + 0 * 16 + ncol, kg, b0);
    load_bfrag(Bt, wn * 32 + 1 * 16 + ncol, kg, b1);
#pragma unroll
    for (int mi = 0; mi < 4; ++mi) {
        acc[mi][0] = __builtin_amdgcn_wmma_f32_16x16x32_bf16(
            false, a[mi].v, false, b0.v, (short)0, acc[mi][0], false, false);
        acc[mi][1] = __builtin_amdgcn_wmma_f32_16x16x32_bf16(
            false, a[mi].v, false, b1.v, (short)0, acc[mi][1], false, false);
    }
}

// --------------------------------------------------------------------------
// Stage 0: unique-occurrence mask per MLP (scatter .set dedupe semantics).
// --------------------------------------------------------------------------
__global__ void prep_mask(const int* __restrict__ idx, unsigned char* __restrict__ mask) {
    int m = blockIdx.x;
    const int* row = idx + m * LSAMP;
    for (int l = threadIdx.x; l < LPAD; l += blockDim.x) {
        unsigned char u = 0;
        if (l < LSAMP) {
            int r = row[l];
            u = 1;
            for (int p = 0; p < l; ++p) {
                if (row[p] == r) { u = 0; break; }
            }
        }
        mask[m * LPAD + l] = u;
    }
}

__global__ void zero_f32(float* __restrict__ p, int n) {
    int i = blockIdx.x * blockDim.x + threadIdx.x;
    if (i < n) p[i] = 0.0f;
}

// --------------------------------------------------------------------------
// GEMM1: h[m][l][n] = gelu( gather(x)[l,:] @ W1[k][:,n] + b1[k][n] ), bf16 out
// grid (ntile=16, ltile=8, m=32), 256 threads (8 waves: 2 L x 4 N),
// software-pipelined (register-staged B tile + A fragments).
// --------------------------------------------------------------------------
__global__ __launch_bounds__(256) void gemm1_kernel(
    const float* __restrict__ x, const int* __restrict__ idx,
    const float* __restrict__ nW1, const float* __restrict__ nb1,
    const float* __restrict__ eW1, const float* __restrict__ eb1,
    __bf16* __restrict__ hbuf)
{
    const int ntile = blockIdx.x;
    const int ltile = blockIdx.y;
    const int m     = blockIdx.z;
    const int kcomp = m & 15;
    const bool isEdge = (m >= 16);

    const float* __restrict__ W = (isEdge ? eW1 : nW1) + (size_t)kcomp * D_ * H_;
    const float* __restrict__ b = (isEdge ? eb1 : nb1) + (size_t)kcomp * H_;

    const int tid  = threadIdx.x;
    const int lane = tid & 31;
    const int wave = tid >> 5;
    const int wl   = wave >> 2;   // 0..1 : 64-row slab
    const int wn   = wave & 3;    // 0..3 : 32-col slab

    __shared__ __align__(16) __bf16 Bt[NB * BST];
    __shared__ __align__(16) __bf16 Cs[LB * CST];

    // Per-lane gathered A-row pointers (WMMA A-layout: lane%16 = row M)
    const int lrow = lane & 15;
    const int ca   = (lane >> 4) * 8;
    const float* arow[4];
    bool avalid[4];
#pragma unroll
    for (int mi = 0; mi < 4; ++mi) {
        int l = ltile * LB + wl * 64 + mi * 16 + lrow;
        if (l < LSAMP) {
            int r   = idx[m * LSAMP + l];
            int t   = r >> 11;
            int tok = (r & 2047) + (isEdge ? (TOK_ / 2) : 0);
            arow[mi]   = x + (size_t)(t * TOK_ + tok) * D_;
            avalid[mi] = true;
        } else {
            arow[mi]   = x;
            avalid[mi] = false;
        }
    }

    v8f zero8;
#pragma unroll
    for (int j = 0; j < 8; ++j) zero8[j] = 0.0f;
    v8f acc[4][2];
#pragma unroll
    for (int mi = 0; mi < 4; ++mi)
#pragma unroll
        for (int ni = 0; ni < 2; ++ni) acc[mi][ni] = zero8;

    const int ncol = lane & 15;
    const int kg   = lane >> 4;
    const int kp   = (tid >> 4) * 2;     // k-pair for B tile staging
    const int hb   = (tid & 15) * 8;     // 8 h-positions per thread

    const int NK = D_ / KB;              // 64 (even)
    BStage bs;
    Frag16 aA[4], aB[4];
    load_btile(W, H_, ntile * NB, 0, kp, hb, bs);
#pragma unroll
    for (int mi = 0; mi < 4; ++mi)
        load_afrag_f32(arow[mi], avalid[mi], 0, ca, aA[mi]);

    for (int kt = 0; kt < NK; kt += 2) {
        // ---- phase 0: compute kt, prefetch kt+1 ----
        __syncthreads();                       // Bt free (prev reads done)
        store_btile(Bt, kp, hb, bs);
        __syncthreads();                       // Bt ready
        load_btile(W, H_, ntile * NB, kt + 1, kp, hb, bs);
#pragma unroll
        for (int mi = 0; mi < 4; ++mi)
            load_afrag_f32(arow[mi], avalid[mi], kt + 1, ca, aB[mi]);
        do_wmma(aA, Bt, wn, ncol, kg, acc);
        // ---- phase 1: compute kt+1, prefetch kt+2 ----
        __syncthreads();
        store_btile(Bt, kp, hb, bs);
        __syncthreads();
        if (kt + 2 < NK) {
            load_btile(W, H_, ntile * NB, kt + 2, kp, hb, bs);
#pragma unroll
            for (int mi = 0; mi < 4; ++mi)
                load_afrag_f32(arow[mi], avalid[mi], kt + 2, ca, aA[mi]);
        }
        do_wmma(aB, Bt, wn, ncol, kg, acc);
    }

    __syncthreads();
    // Epilogue: bias + exact GELU, stage bf16 tile in LDS for coalesced stores
#pragma unroll
    for (int mi = 0; mi < 4; ++mi) {
#pragma unroll
        for (int ni = 0; ni < 2; ++ni) {
            int ccol = wn * 32 + ni * 16 + ncol;
            float bias = b[ntile * NB + ccol];
#pragma unroll
            for (int j = 0; j < 8; ++j) {
                float v = acc[mi][ni][j] + bias;
                v = gelu_exact(v);
                int row = wl * 64 + mi * 16 + kg * 8 + j;
                Cs[row * CST + ccol] = (__bf16)v;
            }
        }
    }
    __syncthreads();
    // 128x128 bf16 tile -> hbuf[m][l][H] row-major, 16B chunks
    for (int e2 = tid; e2 < LB * (NB / 8); e2 += 256) {
        int row = e2 >> 4;
        int col = (e2 & 15) * 8;
        uint4 v = *(const uint4*)&Cs[row * CST + col];
        int l = ltile * LB + row;
        *(uint4*)&hbuf[((size_t)m * LPAD + l) * H_ + ntile * NB + col] = v;
    }
}

// --------------------------------------------------------------------------
// GEMM2: out = h @ W2 + b2, fused masked scatter + segment-sum via atomicAdd
// grid (ntile=8, ltile=8, m=32), 256 threads, same pipelining.
// --------------------------------------------------------------------------
__global__ __launch_bounds__(256) void gemm2_kernel(
    const __bf16* __restrict__ hbuf, const int* __restrict__ idx,
    const unsigned char* __restrict__ mask,
    const float* __restrict__ nW2, const float* __restrict__ nb2,
    const float* __restrict__ eW2, const float* __restrict__ eb2,
    float* __restrict__ pooled)
{
    const int ntile = blockIdx.x;   // 0..7 over C=1024
    const int ltile = blockIdx.y;
    const int m     = blockIdx.z;
    const int kcomp = m & 15;
    const bool isEdge = (m >= 16);

    const float* __restrict__ W = (isEdge ? eW2 : nW2) + (size_t)kcomp * H_ * C_;
    const float* __restrict__ b = (isEdge ? eb2 : nb2) + (size_t)kcomp * C_;

    const int tid  = threadIdx.x;
    const int lane = tid & 31;
    const int wave = tid >> 5;
    const int wl   = wave >> 2;
    const int wn   = wave & 3;

    __shared__ __align__(16) __bf16 Bt[NB * BST];

    const int lrow = lane & 15;
    const int ca   = (lane >> 4) * 8;
    const __bf16* arow[4];
#pragma unroll
    for (int mi = 0; mi < 4; ++mi) {
        int l = ltile * LB + wl * 64 + mi * 16 + lrow;
        arow[mi] = hbuf + ((size_t)m * LPAD + l) * H_;   // padded rows exist
    }

    v8f zero8;
#pragma unroll
    for (int j = 0; j < 8; ++j) zero8[j] = 0.0f;
    v8f acc[4][2];
#pragma unroll
    for (int mi = 0; mi < 4; ++mi)
#pragma unroll
        for (int ni = 0; ni < 2; ++ni) acc[mi][ni] = zero8;

    const int ncol = lane & 15;
    const int kg   = lane >> 4;
    const int kp   = (tid >> 4) * 2;
    const int hb   = (tid & 15) * 8;

    const int NK = H_ / KB;              // 64 (even)
    BStage bs;
    Frag16 aA[4], aB[4];
    load_btile(W, C_, ntile * NB, 0, kp, hb, bs);
#pragma unroll
    for (int mi = 0; mi < 4; ++mi)
        load_afrag_bf16(arow[mi], 0, ca, aA[mi]);

    for (int kt = 0; kt < NK; kt += 2) {
        __syncthreads();
        store_btile(Bt, kp, hb, bs);
        __syncthreads();
        load_btile(W, C_, ntile * NB, kt + 1, kp, hb, bs);
#pragma unroll
        for (int mi = 0; mi < 4; ++mi)
            load_afrag_bf16(arow[mi], kt + 1, ca, aB[mi]);
        do_wmma(aA, Bt, wn, ncol, kg, acc);

        __syncthreads();
        store_btile(Bt, kp, hb, bs);
        __syncthreads();
        if (kt + 2 < NK) {
            load_btile(W, C_, ntile * NB, kt + 2, kp, hb, bs);
#pragma unroll
            for (int mi = 0; mi < 4; ++mi)
                load_afrag_bf16(arow[mi], kt + 2, ca, aA[mi]);
        }
        do_wmma(aB, Bt, wn, ncol, kg, acc);
    }

    // Epilogue: bias + masked scatter-accumulate into pooled[m][t][c]
#pragma unroll
    for (int mi = 0; mi < 4; ++mi) {
#pragma unroll
        for (int j = 0; j < 8; ++j) {
            int l = ltile * LB + wl * 64 + mi * 16 + kg * 8 + j;
            if (l < LSAMP && mask[m * LPAD + l]) {
                int r = idx[m * LSAMP + l];
                int t = r >> 11;
                float* prow = pooled + ((size_t)m * T_ + t) * C_ + ntile * NB;
#pragma unroll
                for (int ni = 0; ni < 2; ++ni) {
                    int c = wn * 32 + ni * 16 + ncol;
                    atomicAdd(prow + c, acc[mi][ni][j] + b[ntile * NB + c]);
                }
            }
        }
    }
}

// --------------------------------------------------------------------------
// Finalize: out[t][m*C + c] = pooled[m][t][c] / TOK
// --------------------------------------------------------------------------
__global__ void finalize_kernel(const float* __restrict__ pooled, float* __restrict__ out) {
    int i = blockIdx.x * blockDim.x + threadIdx.x;
    if (i < T_ * NMLP * C_) {
        int t = i >> 15;
        int rest = i & 32767;
        int m = rest >> 10;
        int c = rest & 1023;
        out[i] = pooled[((size_t)m * T_ + t) * C_ + c] * (1.0f / (float)TOK_);
    }
}

extern "C" void kernel_launch(void* const* d_in, const int* in_sizes, int n_in,
                              void* d_out, int out_size, void* d_ws, size_t ws_size,
                              hipStream_t stream) {
    (void)in_sizes; (void)n_in; (void)out_size; (void)ws_size;
    const float* x    = (const float*)d_in[0];
    const int*   idx  = (const int*)d_in[3];
    const float* nW1  = (const float*)d_in[4];
    const float* nb1  = (const float*)d_in[5];
    const float* nW2  = (const float*)d_in[6];
    const float* nb2  = (const float*)d_in[7];
    const float* eW1  = (const float*)d_in[8];
    const float* eb1  = (const float*)d_in[9];
    const float* eW2  = (const float*)d_in[10];
    const float* eb2  = (const float*)d_in[11];

    unsigned char* mask = (unsigned char*)d_ws;                         // 32 KB
    float*  pooled = (float*)((char*)d_ws + (size_t)(1 << 16));         // 2 MB @ 64KB
    __bf16* hbuf   = (__bf16*)((char*)d_ws + (size_t)(4 << 20));        // 128 MB @ 4MB

    prep_mask<<<NMLP, 256, 0, stream>>>(idx, mask);
    zero_f32<<<(NMLP * T_ * C_ + 255) / 256, 256, 0, stream>>>(pooled, NMLP * T_ * C_);
    gemm1_kernel<<<dim3(H_ / NB, LPAD / LB, NMLP), 256, 0, stream>>>(
        x, idx, nW1, nb1, eW1, eb1, hbuf);
    gemm2_kernel<<<dim3(C_ / NB, LPAD / LB, NMLP), 256, 0, stream>>>(
        hbuf, idx, mask, nW2, nb2, eW2, eb2, pooled);
    finalize_kernel<<<(T_ * NMLP * C_ + 255) / 256, 256, 0, stream>>>(
        pooled, (float*)d_out);
}
